// MambaClassifier_2534030705275
// MI455X (gfx1250) — matrix operand order repaired and checked by reading, not compiled, not run on hardware
//
#include <hip/hip_runtime.h>
#include <hip/hip_bf16.h>
#include <math.h>

// ---------------- problem constants ----------------
#define B_    32
#define L_    4096
#define DM    4
#define DI    64
#define DS    32
#define DC    16
#define TILE  128
#define HALO  (DC - 1)          // 15
#define ROWS  (TILE + HALO)     // 143
#define EPS_  1e-5f

// workspace layout (float offsets)
#define WS_DT  ((size_t)0)                              // [B, L]        dt scalar
#define WS_XC  (WS_DT + (size_t)B_ * L_)                // [B, L, DI]    conv+silu output
#define WS_BM  (WS_XC + (size_t)B_ * L_ * DI)           // [B, L, DS]    B matrix
#define WS_CL  (WS_BM + (size_t)B_ * L_ * DS)           // [B, DS]       C at t=L-1
#define WS_Z   (WS_CL + (size_t)B_ * DS)                // [B, DI]       z at t=L-1
#define WS_YP  (WS_Z  + (size_t)B_ * DI)                // [B, DI]       gated y * fc_w

typedef __attribute__((ext_vector_type(16))) _Float16 v16h;
typedef __attribute__((ext_vector_type(8)))  float    v8f;

__device__ __forceinline__ float silu_f(float v) {
    return v / (1.0f + __expf(-v));
}

// ---- CDNA5 async global->LDS copy (ASYNCcnt pipeline) ----------------
// LDS flat addresses keep the LDS byte offset in the low 32 bits.
__device__ __forceinline__ unsigned lds_off(const void* p) {
    return (unsigned)(unsigned long long)p;
}
__device__ __forceinline__ void async_copy_b128(unsigned lds, unsigned goff,
                                                const void* sbase) {
    asm volatile("global_load_async_to_lds_b128 %0, %1, %2"
                 :: "v"(lds), "v"(goff), "s"(sbase)
                 : "memory");
}
__device__ __forceinline__ void wait_async_all() {
    asm volatile("s_wait_asynccnt 0" ::: "memory");
}

// =====================================================================
// Kernel 1: RMSNorm + in_proj + causal depthwise conv + SiLU + x_proj
// x_proj runs on v_wmma_f32_16x16x32_f16; raw x tile staged via async
// copy; D fragments staged through LDS so global stores are coalesced.
// =====================================================================
#define SM_XN    0u          // float  [ROWS+1][4]      2304 B
#define SM_XIN   2304u       // f16    [ROWS+1][66]    19008 B   (region 0)
#define SM_DBL   0u          // float  [TILE][68]      34816 B   (overlays region 0)
#define SM_XC    34816u      // f16    [TILE][66]      16896 B
#define SM_XRAW  51712u      // float  [ROWS+1][4]      2304 B
#define SM_BYTES 54016u

__global__ void __launch_bounds__(256)
mamba_frontend(const float* __restrict__ x,
               const float* __restrict__ norm_w,
               const float* __restrict__ in_w,     // [2*DI, DM]
               const float* __restrict__ conv_w,   // [DI, 1, DC]
               const float* __restrict__ conv_b,   // [DI]
               const float* __restrict__ xproj_w,  // [1+2*DS, DI] = [65,64]
               float* __restrict__ ws) {
    const int b   = blockIdx.y;
    const int t0  = blockIdx.x * TILE;
    const int tid = threadIdx.x;

    __shared__ __align__(16) unsigned char smem[SM_BYTES];
    float    (*s_xn)[DM]      = (float(*)[DM])(smem + SM_XN);
    _Float16 (*s_xin)[DI + 2] = (_Float16(*)[DI + 2])(smem + SM_XIN);
    float    (*s_dbl)[68]     = (float(*)[68])(smem + SM_DBL);
    _Float16 (*s_xc)[DI + 2]  = (_Float16(*)[DI + 2])(smem + SM_XC);
    float    (*s_xraw)[DM]    = (float(*)[DM])(smem + SM_XRAW);

    float* dtA = ws + WS_DT;
    float* xcA = ws + WS_XC;
    float* bmA = ws + WS_BM;
    float* clA = ws + WS_CL;
    float* zA  = ws + WS_Z;

    // ---- stage raw x rows (tile + halo) into LDS, 16B/row, async ----
    {
        const float* xbase = x + ((size_t)b * L_ + t0 - HALO) * DM;
        if (tid < ROWS) {
            const int t = t0 - HALO + tid;
            if (t >= 0)
                async_copy_b128(lds_off(&s_xraw[tid][0]),
                                (unsigned)(tid * 16), xbase);
        }
        wait_async_all();
        __syncthreads();
    }

    // ---- A) RMSNorm rows -------------------------------------------
    for (int r = tid; r < ROWS; r += blockDim.x) {
        const int t = t0 - HALO + r;
        float v0 = 0.f, v1 = 0.f, v2 = 0.f, v3 = 0.f, rs = 0.f;
        if (t >= 0) {
            v0 = s_xraw[r][0]; v1 = s_xraw[r][1];
            v2 = s_xraw[r][2]; v3 = s_xraw[r][3];
            const float m = (v0*v0 + v1*v1 + v2*v2 + v3*v3) * (1.0f / DM);
            rs = rsqrtf(m + EPS_);
        }
        s_xn[r][0] = v0 * rs * norm_w[0];
        s_xn[r][1] = v1 * rs * norm_w[1];
        s_xn[r][2] = v2 * rs * norm_w[2];
        s_xn[r][3] = v3 * rs * norm_w[3];
    }
    __syncthreads();

    // ---- B) in_proj (x-stream half, K=4 so plain VALU) --------------
    for (int idx = tid; idx < ROWS * DI; idx += blockDim.x) {
        const int r = idx / DI;
        const int d = idx - r * DI;
        const float* w = in_w + d * DM;
        float acc = s_xn[r][0] * w[0];
        acc = fmaf(s_xn[r][1], w[1], acc);
        acc = fmaf(s_xn[r][2], w[2], acc);
        acc = fmaf(s_xn[r][3], w[3], acc);
        s_xin[r][d] = (_Float16)acc;
    }
    // gate z only needed at the final timestep
    if (t0 == L_ - TILE && tid < DI) {
        const float* w = in_w + (DI + tid) * DM;
        float acc = s_xn[ROWS - 1][0] * w[0];
        acc = fmaf(s_xn[ROWS - 1][1], w[1], acc);
        acc = fmaf(s_xn[ROWS - 1][2], w[2], acc);
        acc = fmaf(s_xn[ROWS - 1][3], w[3], acc);
        zA[b * DI + tid] = acc;
    }
    __syncthreads();

    // ---- C) depthwise causal conv-16 + SiLU -------------------------
    for (int idx = tid; idx < TILE * DI; idx += blockDim.x) {
        const int l = idx >> 6;
        const int d = idx & (DI - 1);
        const float* cw = conv_w + d * DC;
        float acc = conv_b[d];
#pragma unroll
        for (int j = 0; j < DC; ++j)
            acc = fmaf(cw[j], (float)s_xin[l + j][d], acc);
        const float xc = silu_f(acc);
        s_xc[l][d] = (_Float16)xc;
        xcA[((size_t)b * L_ + t0 + l) * DI + d] = xc;
    }
    __syncthreads();   // s_xin/s_xn dead from here; s_dbl overlays them

    // ---- D) x_proj GEMM on WMMA: dbl[l,e] = sum_d xc[l,d]*W[e,d] ----
    const int lane = tid & 31;
    const int mrow = lane & 15;       // A: M index / B,D: N index
    const int half = lane >> 4;       // K-subset / M+8 selector
    const int mt   = tid >> 5;        // 8 waves -> 8 M-tiles of 16 rows

    v16h a0, a1;                      // A fragments, K-steps d=0..31 / 32..63
#pragma unroll
    for (int jj = 0; jj < 8; ++jj) {
        const int kb  = ((jj < 4) ? 0 : 16) + half * 8 + 2 * (jj & 3);
        const int row = mt * 16 + mrow;
        a0[2 * jj]     = s_xc[row][kb];
        a0[2 * jj + 1] = s_xc[row][kb + 1];
        a1[2 * jj]     = s_xc[row][32 + kb];
        a1[2 * jj + 1] = s_xc[row][32 + kb + 1];
    }

    for (int nt = 0; nt < 5; ++nt) {
        const int e = nt * 16 + mrow;
        v16h bf0, bf1;
#pragma unroll
        for (int jj = 0; jj < 8; ++jj) {
            const int k = ((jj < 4) ? 0 : 16) + half * 8 + 2 * (jj & 3);
            float w0 = 0.f, w1 = 0.f, w2 = 0.f, w3 = 0.f;
            if (e < 1 + 2 * DS) {     // rows beyond 64 are zero padding
                const float* wr = xproj_w + e * DI;
                w0 = wr[k];      w1 = wr[k + 1];
                w2 = wr[32 + k]; w3 = wr[32 + k + 1];
            }
            bf0[2 * jj] = (_Float16)w0; bf0[2 * jj + 1] = (_Float16)w1;
            bf1[2 * jj] = (_Float16)w2; bf1[2 * jj + 1] = (_Float16)w3;
        }

        v8f acc = {};
        acc = __builtin_amdgcn_wmma_f32_16x16x32_f16(
            false, a0, false, bf0, (short)0, acc, false, false);
        acc = __builtin_amdgcn_wmma_f32_16x16x32_f16(
            false, a1, false, bf1, (short)0, acc, false, false);

        // D layout: lane&15 = N, half -> M+8, vgpr j -> M row. Stage to LDS.
        const int eD = nt * 16 + mrow;
        if (eD <= 2 * DS) {
#pragma unroll
            for (int j = 0; j < 8; ++j)
                s_dbl[mt * 16 + half * 8 + j][eD] = acc[j];
        }
    }
    __syncthreads();

    // ---- E) coalesced stores of dt / B / C(last t) ------------------
    for (int l = tid; l < TILE; l += blockDim.x)
        dtA[(size_t)b * L_ + t0 + l] = s_dbl[l][0];
    for (int idx = tid; idx < TILE * DS; idx += blockDim.x) {
        const int l = idx >> 5;
        const int s = idx & (DS - 1);
        bmA[((size_t)b * L_ + t0 + l) * DS + s] = s_dbl[l][1 + s];
    }
    if (t0 == L_ - TILE && tid < DS)
        clA[b * DS + tid] = s_dbl[TILE - 1][1 + DS + tid];
}

// =====================================================================
// Kernel 2: selective scan. d_state==32 == wave32: one wave per (b,d),
// lane s owns state s. The 8 waves of a block share one batch b, so the
// dt/B streams are staged once per block into LDS with async double
// buffering (global_load_async_to_lds_b128 + s_wait_asynccnt).
// =====================================================================
#define CH   64                 // timesteps per chunk
#define NCH  (L_ / CH)          // 64 chunks

__device__ __forceinline__ void scan_issue_chunk(
        int tid, int c, unsigned bmLds, unsigned xcLds, unsigned dtLds,
        const float* bmBase, const float* xcBase, const float* dtBase) {
    // Bm chunk: CH*DS floats = 8192 B = 512 x b128  (2 ops / thread)
#pragma unroll
    for (int k = 0; k < 2; ++k) {
        const unsigned i = (unsigned)tid + (unsigned)k * 256u;
        async_copy_b128(bmLds + i * 16u,
                        (unsigned)c * (CH * DS * 4u) + i * 16u, bmBase);
    }
    // xc chunk (8 channels): CH*8 floats = 2048 B = 128 x b128
    if (tid < 128) {
        const unsigned t   = (unsigned)tid >> 1;
        const unsigned seg = (unsigned)tid & 1u;
        async_copy_b128(xcLds + t * 32u + seg * 16u,
                        ((unsigned)c * CH + t) * (DI * 4u) + seg * 16u, xcBase);
    } else {
        // dt chunk: CH floats = 256 B = 16 x b128 (extra threads duplicate)
        const unsigned j = (unsigned)tid & 15u;
        async_copy_b128(dtLds + j * 16u,
                        (unsigned)c * (CH * 4u) + j * 16u, dtBase);
    }
}

__global__ void __launch_bounds__(256)
mamba_scan(const float* __restrict__ A_log,
           const float* __restrict__ dt_w,   // [DI, 1]
           const float* __restrict__ dt_b,   // [DI]
           const float* __restrict__ Dp,     // [DI]
           const float* __restrict__ fcw,    // [1, DI]
           float* __restrict__ ws) {
    const int tid  = threadIdx.x;
    const int wave = tid >> 5;
    const int lane = tid & 31;
    const int b    = blockIdx.x >> 3;          // 8 blocks per batch
    const int d0   = (blockIdx.x & 7) << 3;    // 8 channels per block
    const int d    = d0 + wave;

    __shared__ __align__(16) float s_bm[2][CH][DS];  // 16 KB
    __shared__ __align__(16) float s_xcc[2][CH][8];  //  4 KB
    __shared__ __align__(16) float s_dtc[2][CH];     //  0.5 KB

    const float* bmBase = ws + WS_BM + (size_t)b * L_ * DS;
    const float* xcBase = ws + WS_XC + (size_t)b * L_ * DI + d0;
    const float* dtBase = ws + WS_DT + (size_t)b * L_;

    const unsigned bmL0 = lds_off(&s_bm[0][0][0]);
    const unsigned bmL1 = lds_off(&s_bm[1][0][0]);
    const unsigned xcL0 = lds_off(&s_xcc[0][0][0]);
    const unsigned xcL1 = lds_off(&s_xcc[1][0][0]);
    const unsigned dtL0 = lds_off(&s_dtc[0][0]);
    const unsigned dtL1 = lds_off(&s_dtc[1][0]);

    const float As   = -__expf(A_log[d * DS + lane]);
    const float w    = dt_w[d];
    const float bias = dt_b[d];

    scan_issue_chunk(tid, 0, bmL0, xcL0, dtL0, bmBase, xcBase, dtBase);

    float h = 0.f;
    for (int c = 0; c < NCH; ++c) {
        const int buf = c & 1;
        wait_async_all();      // chunk c resident (only its ops outstanding)
        __syncthreads();       // everyone done with chunk c-1's buffer
        if (c + 1 < NCH)       // overlap chunk c+1 load with chunk c compute
            scan_issue_chunk(tid, c + 1,
                             buf ? bmL0 : bmL1, buf ? xcL0 : xcL1,
                             buf ? dtL0 : dtL1, bmBase, xcBase, dtBase);
#pragma unroll 4
        for (int tt = 0; tt < CH; ++tt) {
            const float dtv = s_dtc[buf][tt];
            const float xcv = s_xcc[buf][tt][wave];
            const float bv  = s_bm[buf][tt][lane];
            const float xx  = fmaf(dtv, w, bias);
            const float delta = fmaxf(xx, 0.f) + log1pf(__expf(-fabsf(xx)));
            const float dA = __expf(delta * As);
            h = fmaf(dA, h, delta * xcv * bv);
        }
    }

    // y_{L-1} = h . C_{L-1}  (cross-lane reduction over the 32 states)
    float acc = h * ws[WS_CL + b * DS + lane];
#pragma unroll
    for (int off = 16; off > 0; off >>= 1)
        acc += __shfl_xor(acc, off, 32);

    if (lane == 0) {
        const float xcl = ws[WS_XC + ((size_t)b * L_ + (L_ - 1)) * DI + d];
        float y = acc + Dp[d] * xcl;
        const float zv = ws[WS_Z + b * DI + d];
        y *= silu_f(zv);
        ws[WS_YP + b * DI + d] = y * fcw[d];
    }
}

// =====================================================================
// Kernel 3: head — logits[b] = fc_b + sum_d ypart[b,d]
// =====================================================================
__global__ void mamba_head(const float* __restrict__ fcb,
                           const float* __restrict__ ws,
                           float* __restrict__ out) {
    const int b = threadIdx.x;
    if (b < B_) {
        float acc = fcb[0];
#pragma unroll 8
        for (int d = 0; d < DI; ++d)
            acc += ws[WS_YP + b * DI + d];
        out[b] = acc;
    }
}

// =====================================================================
extern "C" void kernel_launch(void* const* d_in, const int* in_sizes, int n_in,
                              void* d_out, int out_size, void* d_ws, size_t ws_size,
                              hipStream_t stream) {
    (void)in_sizes; (void)n_in; (void)out_size; (void)ws_size;

    const float* x       = (const float*)d_in[0];
    const float* norm_w  = (const float*)d_in[1];
    const float* in_w    = (const float*)d_in[2];
    const float* conv_w  = (const float*)d_in[3];
    const float* conv_b  = (const float*)d_in[4];
    const float* xproj_w = (const float*)d_in[5];
    const float* dt_w    = (const float*)d_in[6];
    const float* dt_b    = (const float*)d_in[7];
    const float* A_log   = (const float*)d_in[8];
    const float* Dp      = (const float*)d_in[9];
    const float* fc_w    = (const float*)d_in[10];
    const float* fc_b    = (const float*)d_in[11];

    float* ws  = (float*)d_ws;
    float* out = (float*)d_out;

    dim3 g1(L_ / TILE, B_);                 // 32 x 32 blocks
    mamba_frontend<<<g1, 256, 0, stream>>>(x, norm_w, in_w, conv_w, conv_b,
                                           xproj_w, ws);
    mamba_scan<<<B_ * (DI / 8), 256, 0, stream>>>(A_log, dt_w, dt_b, Dp,
                                                  fc_w, ws);
    mamba_head<<<1, 32, 0, stream>>>(fc_b, ws, out);
}
